// RnnFamily_386547057206
// MI455X (gfx1250) — matrix operand order repaired and checked by reading, not compiled, not run on hardware
//
#include <hip/hip_runtime.h>
#include <hip/hip_bf16.h>
#include <stdint.h>

typedef __attribute__((ext_vector_type(16))) __bf16 v16bf;
typedef __attribute__((ext_vector_type(8)))  float  v8f;

#define N_NODES_ 500000
#define B_       131072
#define OUT_D_   64

// d_ws layout (bf16 element offsets)
#define OFF_ZR0   0u        // [W0;U0][:, :512]   K=512 N=512
#define OFF_WXH0  262144u   // W0[:, 512:768]     K=256 N=256
#define OFF_UHH0  327680u   // U0[:, 512:768]
#define OFF_ZR1   393216u
#define OFF_WXH1  655360u
#define OFF_UHH1  720896u
#define OFF_WD1   786432u   // Wd1               K=256 N=256
#define OFF_WD2   851968u   // Wd2               K=256 N=64

// ---------------------------------------------------------------------------
// Weight pre-pack: f32 row-major -> bf16 WMMA B-fragment tiles.
// Tile (kt,nt) covers K rows kt*32..+31, N cols nt*16..+15; tile index
// t = kt*(N/16)+nt; within a tile, lane l<16 holds (n=l, K 0..15 pairs),
// lane l+16 holds K 16..31. 512 bf16 (=32B/lane) per tile.
// ---------------------------------------------------------------------------
__global__ __launch_bounds__(256) void pack_b(
    const float* __restrict__ src, int rowStride, int colOff,
    int kBase, int kCount, int N, __bf16* __restrict__ dst)
{
  int idx = blockIdx.x * 256 + threadIdx.x;
  if (idx >= kCount * N) return;
  int kl = idx / N, n = idx - kl * N;
  int k  = kBase + kl;
  int t    = (k >> 5) * (N >> 4) + (n >> 4);
  int kk   = k & 31;
  int lane = (n & 15) + ((kk >> 4) << 4);
  int p    = kk & 15;
  dst[(size_t)t * 512 + lane * 16 + p] =
      (__bf16)src[(size_t)kl * rowStride + colOff + n];
}

// ---------------------------------------------------------------------------
__device__ __forceinline__ float sigf(float x) {
  return 1.0f / (1.0f + __expf(-x));
}
__device__ __forceinline__ float tanhfast(float x) {
  float e = __expf(-2.0f * x);
  return (1.0f - e) / (1.0f + e);
}

// Pack fs (32 rows x 512 f32, row-major) -> A fragments in LDS.
// A tile t = tileBase + rt*ktCount + kt; 16-bit A 16x32 layout:
// lane l<16 : row=l,   K {0..15}; lane l+16: row=l, K {16..31};
// VGPR j holds K pair (2*j [+16 for hi lanes]).
__device__ __forceinline__ void packA_from(
    const float* __restrict__ fs, __bf16* __restrict__ packA,
    int tid, int ktCount, int tileBase, int colBase)
{
  int total = 2 * ktCount * 256;  // bf16 pairs
  for (int p2 = tid; p2 < total; p2 += 256) {
    int tl   = p2 >> 8;               // local tile = rt*ktCount + kt
    int rt   = tl / ktCount, kt = tl - rt * ktCount;
    int lane = (p2 >> 3) & 31;
    int j    = p2 & 7;
    int half = lane >> 4, m = lane & 15;
    int kk0  = 2 * j + 16 * half;
    int row  = rt * 16 + m;
    int k    = colBase + kt * 32 + kk0;
    float f0 = fs[row * 512 + k];
    float f1 = fs[row * 512 + k + 1];
    __bf16* d = packA + (size_t)(tileBase + tl) * 512 + lane * 16 + j * 2;
    d[0] = (__bf16)f0;
    d[1] = (__bf16)f1;
  }
}

// One GRU layer for a 32-row tile.
// fs cols 0..255 = layer input x (overwritten with h_new); cols 256..511 = h.
__device__ __forceinline__ void gru_layer(
    int tid, int wave, int lane,
    const __bf16* __restrict__ wzr, const __bf16* __restrict__ wxh,
    const __bf16* __restrict__ uhh, const float* __restrict__ bias,
    float* fs, __bf16* packA, float* zr,
    float* __restrict__ memOut, const int* sIds, int layerOff)
{
  packA_from(fs, packA, tid, 16, 0, 0);   // [x|h] -> 32 A tiles (K=512)
  __syncthreads();

  const int rt = wave & 1, cg = wave >> 1;
  const int lrow = 8 * (lane >> 4), lcol = lane & 15;

  // ---- z,r gates: [x|h](32x512) @ [W;U][:, :512] -> sigmoid -> zr LDS ----
  {
    v8f acc[8] = {};
    for (int kt = 0; kt < 16; ++kt) {
      v16bf a = *(const v16bf*)(packA + (size_t)(rt * 16 + kt) * 512 + lane * 16);
#pragma unroll
      for (int nt = 0; nt < 8; ++nt) {
        v16bf b = *(const v16bf*)(wzr + (size_t)(kt * 32 + cg * 8 + nt) * 512 + lane * 16);
        acc[nt] = __builtin_amdgcn_wmma_f32_16x16x32_bf16(
            false, a, false, b, (short)0, acc[nt], false, false);
      }
    }
#pragma unroll
    for (int nt = 0; nt < 8; ++nt) {
      int col = (cg * 8 + nt) * 16 + lcol;
      float bb = bias[col];
#pragma unroll
      for (int i = 0; i < 8; ++i) {
        int row = rt * 16 + lrow + i;
        zr[row * 512 + col] = sigf(acc[nt][i] + bb);
      }
    }
  }
  __syncthreads();

  // ---- candidate gate + state update: hh = tanh(gx + b + r*gh) ----
  {
    v8f accx[4] = {}, acch[4] = {};
    for (int kt = 0; kt < 8; ++kt) {
      v16bf ax = *(const v16bf*)(packA + (size_t)(rt * 16 + kt) * 512 + lane * 16);
      v16bf ah = *(const v16bf*)(packA + (size_t)(rt * 16 + 8 + kt) * 512 + lane * 16);
#pragma unroll
      for (int nt = 0; nt < 4; ++nt) {
        int tg = kt * 16 + cg * 4 + nt;
        v16bf bx = *(const v16bf*)(wxh + (size_t)tg * 512 + lane * 16);
        v16bf bh = *(const v16bf*)(uhh + (size_t)tg * 512 + lane * 16);
        accx[nt] = __builtin_amdgcn_wmma_f32_16x16x32_bf16(
            false, ax, false, bx, (short)0, accx[nt], false, false);
        acch[nt] = __builtin_amdgcn_wmma_f32_16x16x32_bf16(
            false, ah, false, bh, (short)0, acch[nt], false, false);
      }
    }
#pragma unroll
    for (int nt = 0; nt < 4; ++nt) {
      int col = (cg * 4 + nt) * 16 + lcol;
      float bb = bias[512 + col];
#pragma unroll
      for (int i = 0; i < 8; ++i) {
        int row = rt * 16 + lrow + i;
        float r  = zr[row * 512 + 256 + col];
        float z  = zr[row * 512 + col];
        float h  = fs[row * 512 + 256 + col];
        float hh = tanhfast(accx[nt][i] + bb + r * acch[nt][i]);
        float hn = z * h + (1.0f - z) * hh;
        fs[row * 512 + col] = hn;                     // feeds next layer
        memOut[(size_t)sIds[row] * 512 + layerOff + col] = hn;  // scatter
      }
    }
  }
  __syncthreads();
}

// ---------------------------------------------------------------------------
__global__ __launch_bounds__(256) void gru_fused(
    const float* __restrict__ node_embs, const float* __restrict__ memory,
    const float* __restrict__ b0, const float* __restrict__ b1,
    const float* __restrict__ bd1, const float* __restrict__ bd2,
    const int* __restrict__ ids,
    const __bf16* __restrict__ wzr0, const __bf16* __restrict__ wxh0,
    const __bf16* __restrict__ uhh0,
    const __bf16* __restrict__ wzr1, const __bf16* __restrict__ wxh1,
    const __bf16* __restrict__ uhh1,
    const __bf16* __restrict__ wd1p, const __bf16* __restrict__ wd2p,
    float* __restrict__ out, float* __restrict__ memOut)
{
  __shared__ float fs[32 * 512];                 // 64KB: [x|h] / activations
  __shared__ float zr[32 * 512];                 // 64KB: z | r gates
  __shared__ float h1f[32 * 256];                // 32KB: layer-1 hidden
  __shared__ __align__(32) __bf16 packA[32 * 512];  // 32KB: A fragments
  __shared__ int sIds[32];

  const int tid = threadIdx.x, wave = tid >> 5, lane = tid & 31;

  if (tid < 32) sIds[tid] = ids[blockIdx.x * 32 + tid];
  __syncthreads();

  // Gather x (cols 0..255), h0 (cols 256..511), h1 -> h1f (coalesced float4)
  for (int q = tid; q < 32 * 64; q += 256) {
    int row = q >> 6, c4 = q & 63;
    const float4* xr = (const float4*)(node_embs + (size_t)sIds[row] * 256);
    const float4* mr = (const float4*)(memory + (size_t)sIds[row] * 512);
    ((float4*)(fs + row * 512))[c4]       = xr[c4];
    ((float4*)(fs + row * 512 + 256))[c4] = mr[c4];
    ((float4*)(h1f + row * 256))[c4]      = mr[64 + c4];
  }
  __syncthreads();

  gru_layer(tid, wave, lane, wzr0, wxh0, uhh0, b0, fs, packA, zr, memOut, sIds, 0);

  // stage layer-2 input: h0_new already in cols 0..255; copy h1 into 256..511
  for (int q = tid; q < 32 * 256; q += 256) {
    int row = q >> 8, c = q & 255;
    fs[row * 512 + 256 + c] = h1f[q];
  }
  __syncthreads();

  gru_layer(tid, wave, lane, wzr1, wxh1, uhh1, b1, fs, packA, zr, memOut, sIds, 256);

  // ---- decoder: relu(h1_new @ Wd1 + bd1) @ Wd2 + bd2 ----
  const int rt = wave & 1, cg = wave >> 1;
  const int lrow = 8 * (lane >> 4), lcol = lane & 15;

  packA_from(fs, packA, tid, 8, 0, 0);     // h1_new (cols 0..255) -> tiles 0..15
  __syncthreads();
  {
    v8f acc[4] = {};
    for (int kt = 0; kt < 8; ++kt) {
      v16bf a = *(const v16bf*)(packA + (size_t)(rt * 8 + kt) * 512 + lane * 16);
#pragma unroll
      for (int nt = 0; nt < 4; ++nt) {
        int tg = kt * 16 + cg * 4 + nt;
        v16bf b = *(const v16bf*)(wd1p + (size_t)tg * 512 + lane * 16);
        acc[nt] = __builtin_amdgcn_wmma_f32_16x16x32_bf16(
            false, a, false, b, (short)0, acc[nt], false, false);
      }
    }
#pragma unroll
    for (int nt = 0; nt < 4; ++nt) {
      int col = (cg * 4 + nt) * 16 + lcol;
      float bb = bd1[col];
#pragma unroll
      for (int i = 0; i < 8; ++i) {
        int row = rt * 16 + lrow + i;
        float v = acc[nt][i] + bb;
        fs[row * 512 + 256 + col] = v > 0.0f ? v : 0.0f;   // d1
      }
    }
  }
  __syncthreads();

  packA_from(fs, packA, tid, 8, 16, 256);  // d1 (cols 256..511) -> tiles 16..31
  __syncthreads();
  {
    v8f acc = {};
    for (int kt = 0; kt < 8; ++kt) {
      v16bf a = *(const v16bf*)(packA + (size_t)(16 + rt * 8 + kt) * 512 + lane * 16);
      v16bf b = *(const v16bf*)(wd2p + (size_t)(kt * 4 + cg) * 512 + lane * 16);
      acc = __builtin_amdgcn_wmma_f32_16x16x32_bf16(
          false, a, false, b, (short)0, acc, false, false);
    }
    float bb = bd2[cg * 16 + lcol];
#pragma unroll
    for (int i = 0; i < 8; ++i) {
      int grow = blockIdx.x * 32 + rt * 16 + lrow + i;
      out[(size_t)grow * 64 + cg * 16 + lcol] = acc[i] + bb;
    }
  }
}

// ---------------------------------------------------------------------------
extern "C" void kernel_launch(void* const* d_in, const int* in_sizes, int n_in,
                              void* d_out, int out_size, void* d_ws, size_t ws_size,
                              hipStream_t stream)
{
  (void)in_sizes; (void)n_in; (void)out_size; (void)ws_size;
  const float* node_embs = (const float*)d_in[0];
  const float* memory    = (const float*)d_in[1];
  const float* W0  = (const float*)d_in[2];
  const float* U0  = (const float*)d_in[3];
  const float* b0  = (const float*)d_in[4];
  const float* W1  = (const float*)d_in[5];
  const float* U1  = (const float*)d_in[6];
  const float* b1  = (const float*)d_in[7];
  const float* Wd1 = (const float*)d_in[8];
  const float* bd1 = (const float*)d_in[9];
  const float* Wd2 = (const float*)d_in[10];
  const float* bd2 = (const float*)d_in[11];
  const int*   ids = (const int*)d_in[12];

  float* out    = (float*)d_out;
  float* memOut = out + (size_t)B_ * OUT_D_;

  __bf16* ws = (__bf16*)d_ws;
  __bf16 *wzr0 = ws + OFF_ZR0, *wxh0 = ws + OFF_WXH0, *uhh0 = ws + OFF_UHH0;
  __bf16 *wzr1 = ws + OFF_ZR1, *wxh1 = ws + OFF_WXH1, *uhh1 = ws + OFF_UHH1;
  __bf16 *wd1p = ws + OFF_WD1, *wd2p = ws + OFF_WD2;

  // Weight pre-pack (f32 -> bf16 fragment tiles); tiny, rewritten every call.
  #define PACK(src, rs, co, kb, kc, N, dst) \
    pack_b<<<((kc) * (N) + 255) / 256, 256, 0, stream>>>(src, rs, co, kb, kc, N, dst)
  PACK(W0, 768, 0, 0, 256, 512, wzr0);
  PACK(U0, 768, 0, 256, 256, 512, wzr0);
  PACK(W0, 768, 512, 0, 256, 256, wxh0);
  PACK(U0, 768, 512, 0, 256, 256, uhh0);
  PACK(W1, 768, 0, 0, 256, 512, wzr1);
  PACK(U1, 768, 0, 256, 256, 512, wzr1);
  PACK(W1, 768, 512, 0, 256, 256, wxh1);
  PACK(U1, 768, 512, 0, 256, 256, uhh1);
  PACK(Wd1, 256, 0, 0, 256, 256, wd1p);
  PACK(Wd2, 64, 0, 0, 256, 64, wd2p);
  #undef PACK

  // Bulk new_memory = memory (1 GB, HBM-bound), then fused kernel scatters
  // the updated rows over it.
  hipMemcpyAsync(memOut, memory, (size_t)N_NODES_ * 512 * sizeof(float),
                 hipMemcpyDeviceToDevice, stream);

  gru_fused<<<B_ / 32, 256, 0, stream>>>(
      node_embs, memory, b0, b1, bd1, bd2, ids,
      wzr0, wxh0, uhh0, wzr1, wxh1, uhh1, wd1p, wd2p, out, memOut);
}
